// CausalSelfAttention_59966333386984
// MI455X (gfx1250) — matrix operand order repaired and checked by reading, not compiled, not run on hardware
//
#include <hip/hip_runtime.h>
#include <hip/hip_bf16.h>
#include <cstddef>

// ---------------------------------------------------------------------------
// CDNA5 (gfx1250) causal self-attention forward, bf16 WMMA pipeline.
// wave32 everywhere; one wave per workgroup for the matrix kernels.
// ---------------------------------------------------------------------------

typedef __attribute__((ext_vector_type(16))) __bf16 bf16x16;
typedef __attribute__((ext_vector_type(8)))  __bf16 bf16x8;
typedef __attribute__((ext_vector_type(8)))  float  f32x8;

#define DEVINL __device__ __forceinline__

constexpr int BB = 4, TT = 2048, CC = 1024, NH = 16, HS = 64;

// workspace layout (bf16 element offsets)
constexpr size_t XB_OFF  = 0;                                   // x bf16      [8192,1024]
constexpr size_t WAB_OFF = XB_OFF  + (size_t)BB * TT * CC;      // w_attn bf16 [3072,1024]
constexpr size_t WPB_OFF = WAB_OFF + (size_t)3 * CC * CC;       // w_proj bf16 [1024,1024]
constexpr size_t QB_OFF  = WPB_OFF + (size_t)CC * CC;           // Q  [64,2048,64]
constexpr size_t KB_OFF  = QB_OFF  + (size_t)BB * NH * TT * HS; // K  [64,2048,64]
constexpr size_t VT_OFF  = KB_OFF  + (size_t)BB * NH * TT * HS; // Vt [64,64,2048]
constexpr size_t YB_OFF  = VT_OFF  + (size_t)BB * NH * HS * TT; // y  [8192,1024]

DEVINL f32x8 wmma_bf16(bf16x16 a, bf16x16 b, f32x8 c) {
  // D = A(16x32) * B(32x16) + C(16x16 f32)
  return __builtin_amdgcn_wmma_f32_16x16x32_bf16(
      /*neg_a=*/false, a, /*neg_b=*/false, b,
      /*c_mod=*/(short)0, c, /*reuse_a=*/false, /*reuse_b=*/false);
}

// A-matrix fragment, row-major source [row, k], ld in elements.
// Layout used (contiguous k, as in frag_b): lane(0-15)=rows, half=lane>>4; elem e -> K = k0+16*kh+e.
DEVINL bf16x16 frag_a(const __bf16* __restrict__ base, int ld, int row0, int k0) {
  const int tid = threadIdx.x & 31;
  const int r   = tid & 15;
  const int kh  = tid >> 4;
  const __bf16* p = base + (size_t)(row0 + r) * ld + k0 + 16 * kh;
  bf16x8 lo = *(const bf16x8*)(p);
  bf16x8 hi = *(const bf16x8*)(p + 8);
  bf16x16 v;
#pragma unroll
  for (int e = 0; e < 8; ++e) { v[e] = lo[e]; v[e + 8] = hi[e]; }
  return v;
}

// B-matrix fragment. Source holds B-columns as contiguous rows: src[n, k].
// Layout used (contiguous k, as in frag_a): lane(0-15)=cols, elem e -> K = k0+16*kh+e.
DEVINL bf16x16 frag_b(const __bf16* __restrict__ base, int ld, int n0, int k0) {
  const int tid = threadIdx.x & 31;
  const int n   = tid & 15;
  const int kh  = tid >> 4;
  const __bf16* p = base + (size_t)(n0 + n) * ld + k0 + 16 * kh;
  bf16x8 lo = *(const bf16x8*)(p);
  bf16x8 hi = *(const bf16x8*)(p + 8);
  bf16x16 v;
#pragma unroll
  for (int e = 0; e < 8; ++e) { v[e] = lo[e]; v[e + 8] = hi[e]; }
  return v;
}

// ---------------------------------------------------------------------------
// fp32 -> bf16 convert, 8 elems/lane, b128 in / b128 out
// ---------------------------------------------------------------------------
__global__ void __launch_bounds__(256)
f32_to_bf16_kernel(const float* __restrict__ in, __bf16* __restrict__ out, int n) {
  int idx = (blockIdx.x * 256 + threadIdx.x) * 8;
  if (idx >= n) return;
  float4 a = ((const float4*)(in + idx))[0];
  float4 b = ((const float4*)(in + idx))[1];
  bf16x8 v;
  v[0] = (__bf16)a.x; v[1] = (__bf16)a.y; v[2] = (__bf16)a.z; v[3] = (__bf16)a.w;
  v[4] = (__bf16)b.x; v[5] = (__bf16)b.y; v[6] = (__bf16)b.z; v[7] = (__bf16)b.w;
  *(bf16x8*)(out + idx) = v;
}

// ---------------------------------------------------------------------------
// 64x64-per-wave GEMM:  out[m,n] = sum_k A[m,k] * W[n,k]   (K = 1024)
// EPI 0: scatter to Q / K / V^T (bf16).  EPI 1: plain f32 store.
// ---------------------------------------------------------------------------
template <int EPI>
__global__ void __launch_bounds__(32)
gemm64_kernel(const __bf16* __restrict__ A, const __bf16* __restrict__ W, int Kdim,
              __bf16* __restrict__ qb, __bf16* __restrict__ kb,
              __bf16* __restrict__ vtb, float* __restrict__ out) {
  const int row0 = blockIdx.y * 64;
  const int col0 = blockIdx.x * 64;
  const int lane = threadIdx.x & 31;
  const int l15  = lane & 15;
  const int half = lane >> 4;

  f32x8 acc[4][4] = {};

  for (int k0 = 0; k0 < Kdim; k0 += 32) {
    bf16x16 af[4], bfr[4];
#pragma unroll
    for (int i = 0; i < 4; ++i) af[i]  = frag_a(A, Kdim, row0 + 16 * i, k0);
#pragma unroll
    for (int j = 0; j < 4; ++j) bfr[j] = frag_b(W, Kdim, col0 + 16 * j, k0);
#pragma unroll
    for (int i = 0; i < 4; ++i)
#pragma unroll
      for (int j = 0; j < 4; ++j)
        acc[i][j] = wmma_bf16(af[i], bfr[j], acc[i][j]);
  }

  if (EPI == 1) {
#pragma unroll
    for (int i = 0; i < 4; ++i)
#pragma unroll
      for (int j = 0; j < 4; ++j)
#pragma unroll
        for (int r = 0; r < 8; ++r) {
          int m = row0 + 16 * i + r + 8 * half;
          int n = col0 + 16 * j + l15;
          out[(size_t)m * CC + n] = acc[i][j][r];
        }
  } else {
    const int seg = col0 >> 10;  // uniform per block (tiles never straddle 1024)
#pragma unroll
    for (int i = 0; i < 4; ++i)
#pragma unroll
      for (int j = 0; j < 4; ++j)
#pragma unroll
        for (int r = 0; r < 8; ++r) {
          int m = row0 + 16 * i + r + 8 * half;   // token index 0..8191
          int n = col0 + 16 * j + l15;            // 0..3071
          int b = m >> 11, t = m & 2047;
          int h = (n >> 6) & 15;
          int d = n & 63;
          int bh = b * NH + h;
          __bf16 val = (__bf16)acc[i][j][r];
          if (seg == 0)      qb[((size_t)bh * TT + t) * HS + d] = val;
          else if (seg == 1) kb[((size_t)bh * TT + t) * HS + d] = val;
          else               vtb[((size_t)bh * HS + d) * TT + t] = val;  // V^T
        }
  }
}

// ---------------------------------------------------------------------------
// Flash attention: one wave handles (head bh, 32 query rows).
// S = Q K^T (WMMA bf16), online softmax in f32, O += P V (WMMA bf16, V^T src).
// ---------------------------------------------------------------------------
__global__ void __launch_bounds__(32)
attn_kernel(const __bf16* __restrict__ qb, const __bf16* __restrict__ kb,
            const __bf16* __restrict__ vtb, __bf16* __restrict__ yb) {
  __shared__ __align__(16) __bf16 ps[32 * 32];  // P tile staging (2 KB)

  const int lane = threadIdx.x & 31;
  const int l15  = lane & 15;
  const int half = lane >> 4;
  const int bh   = blockIdx.y;
  const int q0   = blockIdx.x * 32;
  const float SCALE = 0.125f;  // 1/sqrt(64)

  const __bf16* Q  = qb  + (size_t)bh * TT * HS;
  const __bf16* Kp = kb  + (size_t)bh * TT * HS;
  const __bf16* Vt = vtb + (size_t)bh * HS * TT;

  // Q fragments held in registers for the whole pass
  bf16x16 qf[2][2];
#pragma unroll
  for (int i = 0; i < 2; ++i)
#pragma unroll
    for (int kk = 0; kk < 2; ++kk)
      qf[i][kk] = frag_a(Q, HS, q0 + 16 * i, 32 * kk);

  f32x8 o[2][4] = {};
  float mrow[2][8], lrow[2][8];
#pragma unroll
  for (int i = 0; i < 2; ++i)
#pragma unroll
    for (int r = 0; r < 8; ++r) { mrow[i][r] = -1e30f; lrow[i][r] = 0.0f; }

  const int ntiles = blockIdx.x + 1;  // causal: kv tiles 0 .. diag
  for (int tkv = 0; tkv < ntiles; ++tkv) {
    const int kv0 = tkv * 32;

    // ---- S = Q * K^T  (32 q-rows x 32 kv-cols) ----
    f32x8 s[2][2] = {};
#pragma unroll
    for (int kk = 0; kk < 2; ++kk)
#pragma unroll
      for (int j = 0; j < 2; ++j) {
        bf16x16 kfrag = frag_b(Kp, HS, kv0 + 16 * j, 32 * kk);
#pragma unroll
        for (int i = 0; i < 2; ++i)
          s[i][j] = wmma_bf16(qf[i][kk], kfrag, s[i][j]);
      }

    // ---- mask + online softmax ----
#pragma unroll
    for (int i = 0; i < 2; ++i)
#pragma unroll
      for (int r = 0; r < 8; ++r) {
        const int row = q0 + 16 * i + r + 8 * half;
        float v0 = s[i][0][r] * SCALE;
        float v1 = s[i][1][r] * SCALE;
        if (kv0 + l15 > row)      v0 = -1e30f;
        if (kv0 + 16 + l15 > row) v1 = -1e30f;

        float t = fmaxf(v0, v1);
        t = fmaxf(t, __shfl_xor(t, 1, 32));
        t = fmaxf(t, __shfl_xor(t, 2, 32));
        t = fmaxf(t, __shfl_xor(t, 4, 32));
        t = fmaxf(t, __shfl_xor(t, 8, 32));   // row max (within 16-lane half)

        const float mo = mrow[i][r];
        const float mn = fmaxf(mo, t);
        const float sc = __expf(mo - mn);
        const float p0 = __expf(v0 - mn);
        const float p1 = __expf(v1 - mn);
        float rs = p0 + p1;
        rs += __shfl_xor(rs, 1, 32);
        rs += __shfl_xor(rs, 2, 32);
        rs += __shfl_xor(rs, 4, 32);
        rs += __shfl_xor(rs, 8, 32);          // row sum

        lrow[i][r] = lrow[i][r] * sc + rs;
        mrow[i][r] = mn;
#pragma unroll
        for (int nt = 0; nt < 4; ++nt) o[i][nt][r] *= sc;

        const int lr = 16 * i + r + 8 * half;  // local row in P tile
        ps[lr * 32 + l15]      = (__bf16)p0;
        ps[lr * 32 + 16 + l15] = (__bf16)p1;
      }

    asm volatile("s_wait_dscnt 0" ::: "memory");

    // ---- O += P * V  (P re-layout via LDS, V from transposed layout) ----
    bf16x16 pf[2];
#pragma unroll
    for (int i = 0; i < 2; ++i) pf[i] = frag_a(ps, 32, 16 * i, 0);

#pragma unroll
    for (int nt = 0; nt < 4; ++nt) {
      bf16x16 vfrag = frag_b(Vt, TT, 16 * nt, kv0);
#pragma unroll
      for (int i = 0; i < 2; ++i)
        o[i][nt] = wmma_bf16(pf[i], vfrag, o[i][nt]);
    }
    asm volatile("s_wait_dscnt 0" ::: "memory");
  }

  // ---- normalize and store y[b, t, h*64 + d] (bf16) ----
  const int b = bh >> 4, h = bh & 15;
#pragma unroll
  for (int i = 0; i < 2; ++i)
#pragma unroll
    for (int nt = 0; nt < 4; ++nt)
#pragma unroll
      for (int r = 0; r < 8; ++r) {
        const int t = q0 + 16 * i + r + 8 * half;
        const float val = o[i][nt][r] / lrow[i][r];
        yb[((size_t)(b * TT + t)) * CC + h * HS + 16 * nt + l15] = (__bf16)val;
      }
}

// ---------------------------------------------------------------------------
// launch
// ---------------------------------------------------------------------------
extern "C" void kernel_launch(void* const* d_in, const int* in_sizes, int n_in,
                              void* d_out, int out_size, void* d_ws, size_t ws_size,
                              hipStream_t stream) {
  const float* x  = (const float*)d_in[0];   // [4,2048,1024]
  const float* wa = (const float*)d_in[1];   // [3072,1024]
  const float* wp = (const float*)d_in[2];   // [1024,1024]
  float* out = (float*)d_out;                // [4,2048,1024]
  __bf16* ws = (__bf16*)d_ws;

  const int nx = BB * TT * CC;       // 8388608
  const int na = 3 * CC * CC;        // 3145728
  const int np = CC * CC;            // 1048576

  f32_to_bf16_kernel<<<nx / (8 * 256), 256, 0, stream>>>(x,  ws + XB_OFF,  nx);
  f32_to_bf16_kernel<<<na / (8 * 256), 256, 0, stream>>>(wa, ws + WAB_OFF, na);
  f32_to_bf16_kernel<<<np / (8 * 256), 256, 0, stream>>>(wp, ws + WPB_OFF, np);

  // QKV: [8192,3072] = x_bf16 @ w_attn^T, scattered to Q/K/V^T
  gemm64_kernel<0><<<dim3(3 * CC / 64, BB * TT / 64), 32, 0, stream>>>(
      ws + XB_OFF, ws + WAB_OFF, CC,
      ws + QB_OFF, ws + KB_OFF, ws + VT_OFF, nullptr);

  // flash attention per (32 q-rows, head)
  attn_kernel<<<dim3(TT / 32, BB * NH), 32, 0, stream>>>(
      ws + QB_OFF, ws + KB_OFF, ws + VT_OFF, ws + YB_OFF);

  // proj: out = y_bf16 @ w_proj^T  (fp32 out)
  gemm64_kernel<1><<<dim3(CC / 64, BB * TT / 64), 32, 0, stream>>>(
      ws + YB_OFF, ws + WPB_OFF, CC,
      nullptr, nullptr, nullptr, out);
}
